// PolymerGNN_jointMono_SinglePool_85469849191001
// MI455X (gfx1250) — compile-verified, matrix operand order checked by reading, not run on hardware
//
#include <hip/hip_runtime.h>

#define Nn 50000
#define Ee 640000
#define Hh 128
#define Kk 25000

typedef __attribute__((ext_vector_type(16))) __bf16 v16bf;
typedef __attribute__((ext_vector_type(8)))  float  v8f;

__device__ __forceinline__ unsigned flipF(float f) {
    unsigned u = __float_as_uint(f);
    return (u & 0x80000000u) ? ~u : (u | 0x80000000u);
}
__device__ __forceinline__ float unflipF(unsigned u) {
    return (u & 0x80000000u) ? __uint_as_float(u ^ 0x80000000u)
                             : __uint_as_float(~u);
}

// ---------------- weight pre-pack into WMMA B-fragment order (bf16) ----------------
// Fragment index t = (w*4 + ks)*32 + lane; 16 bf16 per fragment-lane.
// Element (i<8):  W[ks*32 + hh*8 + i      , w*16 + (lane&15)]
// Element (i>=8): W[ks*32 + 16 + hh*8 + i-8, w*16 + (lane&15)]
__global__ __launch_bounds__(256) void pack_b(const float* __restrict__ W,
                                              __bf16* __restrict__ Bp)
{
    int t = blockIdx.x * 256 + threadIdx.x;      // 0..1023
    int w    = t >> 7;
    int ks   = (t >> 5) & 3;
    int lane = t & 31;
    int l = lane & 15, hh = lane >> 4;
    int kb = ks * 32;
    __bf16* out = Bp + (size_t)t * 16;
#pragma unroll
    for (int i = 0; i < 8; ++i) {
        out[i]     = (__bf16)W[(size_t)(kb + hh * 8 + i) * 128 + w * 16 + l];
        out[i + 8] = (__bf16)W[(size_t)(kb + 16 + hh * 8 + i) * 128 + w * 16 + l];
    }
}

// ---------------- WMMA GEMM: C[N,128] = A1@B1 (+ A2@B2) + bias ----------------
// Block = 256 threads = 8 waves; block covers 64 rows x 128 cols.
// Wave w -> cols [16w,16w+16), 4 M-tiles of 16 rows sharing one B fragment per k-step.
__global__ __launch_bounds__(256) void gemm_h128(
    const float* __restrict__ A1, const __bf16* __restrict__ B1p,
    const float* __restrict__ A2, const __bf16* __restrict__ B2p,
    const float* __restrict__ bias, float* __restrict__ C, int nrows)
{
    int wave = threadIdx.x >> 5;
    int lane = threadIdx.x & 31;
    int l  = lane & 15;          // M for A, N for B/C
    int hh = lane >> 4;          // K-half selector
    int row0 = blockIdx.x * 64;
    int col0 = wave * 16;

    v8f acc[4] = {};
    const v16bf* B1v = (const v16bf*)B1p + (size_t)wave * 4 * 32 + lane;
    const v16bf* B2v = A2 ? ((const v16bf*)B2p + (size_t)wave * 4 * 32 + lane) : nullptr;

    int rowm[4];
#pragma unroll
    for (int m = 0; m < 4; ++m) {
        int r = row0 + m * 16 + l;
        rowm[m] = (r < nrows) ? r : (nrows - 1);   // clamp: masked at store
    }

    for (int ks = 0; ks < 4; ++ks) {
        int kb = ks * 32;
        v16bf b1 = B1v[(size_t)ks * 32];
#pragma unroll
        for (int m = 0; m < 4; ++m) {
            const float* ar = A1 + (size_t)rowm[m] * 128 + kb;
            v16bf a;
#pragma unroll
            for (int i = 0; i < 8; ++i) {
                a[i]     = (__bf16)ar[hh * 8 + i];
                a[i + 8] = (__bf16)ar[16 + hh * 8 + i];
            }
            acc[m] = __builtin_amdgcn_wmma_f32_16x16x32_bf16(false, a, false, b1,
                                                             (short)0, acc[m], false, false);
        }
        if (A2) {
            v16bf b2 = B2v[(size_t)ks * 32];
#pragma unroll
            for (int m = 0; m < 4; ++m) {
                const float* ar = A2 + (size_t)rowm[m] * 128 + kb;
                v16bf a;
#pragma unroll
                for (int i = 0; i < 8; ++i) {
                    a[i]     = (__bf16)ar[hh * 8 + i];
                    a[i + 8] = (__bf16)ar[16 + hh * 8 + i];
                }
                acc[m] = __builtin_amdgcn_wmma_f32_16x16x32_bf16(false, a, false, b2,
                                                                 (short)0, acc[m], false, false);
            }
        }
    }
    float bc = bias ? bias[col0 + l] : 0.0f;
#pragma unroll
    for (int m = 0; m < 4; ++m) {
#pragma unroll
        for (int v = 0; v < 8; ++v) {
            int r = row0 + m * 16 + v + 8 * hh;
            if (r < nrows) C[(size_t)r * 128 + col0 + l] = acc[m][v] + bc;
        }
    }
}

// ---------------- small fills ----------------
__global__ void fill_u32(unsigned* p, unsigned v, size_t n) {
    size_t i = (size_t)blockIdx.x * blockDim.x + threadIdx.x;
    if (i < n) p[i] = v;
}
__global__ void fill_f32(float* p, float v, size_t n) {
    size_t i = (size_t)blockIdx.x * blockDim.x + threadIdx.x;
    if (i < n) p[i] = v;
}

// ---------------- GAT attention projections: s = xw @ a ----------------
__global__ __launch_bounds__(256) void gat_proj(
    const float* __restrict__ xw, const float* __restrict__ asrc,
    const float* __restrict__ adst, float* __restrict__ ssrc,
    float* __restrict__ sdst, int n)
{
    int node = blockIdx.x * 8 + (threadIdx.x >> 5);
    int lane = threadIdx.x & 31;
    if (node >= n) return;
    const float* row = xw + (size_t)node * 128;
    float s0 = 0.f, s1 = 0.f;
    for (int i = lane; i < 128; i += 32) {
        float v = row[i];
        s0 += v * asrc[i];
        s1 += v * adst[i];
    }
    for (int m = 16; m > 0; m >>= 1) {
        s0 += __shfl_xor(s0, m, 32);
        s1 += __shfl_xor(s1, m, 32);
    }
    if (lane == 0) { ssrc[node] = s0; sdst[node] = s1; }
}

// ---------------- GAT softmax pieces (edges incl. self loops) ----------------
__global__ void gat_edge_max(const int* __restrict__ src, const int* __restrict__ dst,
                             int E, int n, const float* __restrict__ ssrc,
                             const float* __restrict__ sdst, unsigned* __restrict__ menc)
{
    int i = blockIdx.x * blockDim.x + threadIdx.x;
    if (i >= E + n) return;
    int s, d;
    if (i < E) { s = src[i]; d = dst[i]; } else { s = i - E; d = s; }
    float e = ssrc[s] + sdst[d];
    e = (e >= 0.f) ? e : 0.2f * e;
    atomicMax(menc + d, flipF(e));
}

__global__ void unflip_inplace(unsigned* p, size_t n) {
    size_t i = (size_t)blockIdx.x * blockDim.x + threadIdx.x;
    if (i < n) p[i] = __float_as_uint(unflipF(p[i]));
}

__global__ void gat_edge_den(const int* __restrict__ src, const int* __restrict__ dst,
                             int E, int n, const float* __restrict__ ssrc,
                             const float* __restrict__ sdst,
                             const float* __restrict__ mrow, float* __restrict__ denom)
{
    int i = blockIdx.x * blockDim.x + threadIdx.x;
    if (i >= E + n) return;
    int s, d;
    if (i < E) { s = src[i]; d = dst[i]; } else { s = i - E; d = s; }
    float e = ssrc[s] + sdst[d];
    e = (e >= 0.f) ? e : 0.2f * e;
    atomicAdd(denom + d, expf(e - mrow[d]));
}

// segment_max over edges of alpha * xw[src] -> aggenc (one (edge,4-feature) per thread)
__global__ void gat_edge_aggmax(const int* __restrict__ src, const int* __restrict__ dst,
                                int E, int n, const float* __restrict__ ssrc,
                                const float* __restrict__ sdst,
                                const float* __restrict__ mrow,
                                const float* __restrict__ denom,
                                const float* __restrict__ xw, unsigned* __restrict__ aggenc)
{
    long t = (long)blockIdx.x * blockDim.x + threadIdx.x;
    long tot = (long)(E + n) * 32;
    if (t >= tot) return;
    int i = (int)(t >> 5), lane = (int)(t & 31);
    int s, d;
    if (i < E) { s = src[i]; d = dst[i]; } else { s = i - E; d = s; }
    float e = ssrc[s] + sdst[d];
    e = (e >= 0.f) ? e : 0.2f * e;
    float alpha = expf(e - mrow[d]) / denom[d];
    const float* xr = xw + (size_t)s * 128 + lane * 4;
    unsigned* out = aggenc + (size_t)d * 128 + lane * 4;
#pragma unroll
    for (int j = 0; j < 4; ++j) atomicMax(out + j, flipF(alpha * xr[j]));
}

__global__ void decode_bias(const unsigned* __restrict__ enc,
                            const float* __restrict__ bias,
                            float* __restrict__ out, size_t tot)
{
    size_t i = (size_t)blockIdx.x * blockDim.x + threadIdx.x;
    if (i >= tot) return;
    out[i] = unflipF(enc[i]) + bias[i & 127];
}

// ---------------- BatchNorm (batch stats) + PReLU ----------------
__global__ __launch_bounds__(128) void bn_stats(const float* __restrict__ x, int n,
                                                float* __restrict__ sum,
                                                float* __restrict__ sumsq)
{
    int f = threadIdx.x;
    int r0 = blockIdx.x * 256;
    int r1 = r0 + 256; if (r1 > n) r1 = n;
    float s = 0.f, q = 0.f;
    for (int r = r0; r < r1; ++r) {
        float v = x[(size_t)r * 128 + f];
        s += v; q += v * v;
    }
    atomicAdd(sum + f, s);
    atomicAdd(sumsq + f, q);
}

__global__ void bn_prelu(const float* __restrict__ x, float* __restrict__ y, int n,
                         const float* __restrict__ sum, const float* __restrict__ sumsq,
                         const float* __restrict__ w, const float* __restrict__ b,
                         const float* __restrict__ pa)
{
    size_t i = (size_t)blockIdx.x * blockDim.x + threadIdx.x;
    size_t tot = (size_t)n * 128;
    if (i >= tot) return;
    int f = (int)(i & 127);
    float inv_n = 1.0f / (float)n;
    float mu = sum[f] * inv_n;
    float var = sumsq[f] * inv_n - mu * mu;
    float v = (x[i] - mu) * rsqrtf(var + 1e-5f) * w[f] + b[f];
    float a = pa[0];
    y[i] = (v >= 0.f) ? v : a * v;
}

// ---------------- SAGE segment_max / pool segment_sum ----------------
__global__ void edge_feat_max(const int* __restrict__ src, const int* __restrict__ dst,
                              int E, const float* __restrict__ h, unsigned* __restrict__ enc)
{
    long t = (long)blockIdx.x * blockDim.x + threadIdx.x;
    long tot = (long)E * 32;
    if (t >= tot) return;
    int i = (int)(t >> 5), lane = (int)(t & 31);
    int s = src[i], d = dst[i];
    const float* xr = h + (size_t)s * 128 + lane * 4;
    unsigned* out = enc + (size_t)d * 128 + lane * 4;
#pragma unroll
    for (int j = 0; j < 4; ++j) atomicMax(out + j, flipF(xr[j]));
}

__global__ void sage_decode(unsigned* __restrict__ enc, size_t tot) {
    size_t i = (size_t)blockIdx.x * blockDim.x + threadIdx.x;
    if (i >= tot) return;
    unsigned u = enc[i];
    float v = (u == 0u) ? 0.0f : unflipF(u);        // isolated node -> 0
    if (!isfinite(v)) v = 0.0f;
    enc[i] = __float_as_uint(v);
}

__global__ void edge_feat_sum(const int* __restrict__ src, const int* __restrict__ dst,
                              int E, const float* __restrict__ h, float* __restrict__ acc)
{
    long t = (long)blockIdx.x * blockDim.x + threadIdx.x;
    long tot = (long)E * 32;
    if (t >= tot) return;
    int i = (int)(t >> 5), lane = (int)(t & 31);
    int s = src[i], d = dst[i];
    const float* xr = h + (size_t)s * 128 + lane * 4;
    float* out = acc + (size_t)d * 128 + lane * 4;
#pragma unroll
    for (int j = 0; j < 4; ++j) atomicAdd(out + j, xr[j]);
}

// ---------------- SAGPool score ----------------
__global__ __launch_bounds__(256) void pool_score(
    const float* __restrict__ agg, const float* __restrict__ h,
    const float* __restrict__ wrel, const float* __restrict__ wroot,
    const float* __restrict__ brel, float* __restrict__ score,
    unsigned* __restrict__ keys, int n)
{
    int node = blockIdx.x * 8 + (threadIdx.x >> 5);
    int lane = threadIdx.x & 31;
    if (node >= n) return;
    const float* ar = agg + (size_t)node * 128;
    const float* hr = h + (size_t)node * 128;
    float s = 0.f;
    for (int i = lane; i < 128; i += 32) s += ar[i] * wrel[i] + hr[i] * wroot[i];
    for (int m = 16; m > 0; m >>= 1) s += __shfl_xor(s, m, 32);
    if (lane == 0) {
        float sc = s + brel[0];
        score[node] = sc;
        keys[node] = flipF(sc);
    }
}

// ---------------- MSB radix select of K-th largest key ----------------
__global__ void init_state(int* state, int K) {
    if (threadIdx.x == 0 && blockIdx.x == 0) { state[0] = 0; state[1] = K; state[2] = 0; }
}

__global__ __launch_bounds__(256) void radix_hist(const unsigned* __restrict__ keys, int n,
                                                  const int* __restrict__ state, int shift,
                                                  unsigned* __restrict__ hist)
{
    __shared__ unsigned sh[256];
    sh[threadIdx.x] = 0u;
    __syncthreads();
    int i = blockIdx.x * blockDim.x + threadIdx.x;
    if (i < n) {
        unsigned k = keys[i];
        unsigned prefix = (unsigned)state[0];
        unsigned hm = (shift >= 24) ? 0u : (0xFFFFFFFFu << (shift + 8));
        if ((k & hm) == (prefix & hm)) atomicAdd(&sh[(k >> shift) & 255], 1u);
    }
    __syncthreads();
    if (sh[threadIdx.x]) atomicAdd(hist + threadIdx.x, sh[threadIdx.x]);
}

__global__ void radix_select(const unsigned* __restrict__ hist, int* state,
                             int shift, int lastpass)
{
    if (threadIdx.x != 0 || blockIdx.x != 0) return;
    int krem = state[1];
    unsigned prefix = (unsigned)state[0];
    int b = 255;
    for (; b > 0; --b) {
        int c = (int)hist[b];
        if (c >= krem) break;
        krem -= c;
    }
    state[0] = (int)(prefix | ((unsigned)b << shift));
    state[1] = krem;
    if (lastpass) state[2] = 0;   // reset tie counter for the gather pass
}

// ---------------- gather top-K, gate by tanh(score), column-max into agenc ----------------
__global__ __launch_bounds__(256) void pool_gather(
    const unsigned* __restrict__ keys, const float* __restrict__ score,
    const float* __restrict__ h, int n, int* state, unsigned* __restrict__ agenc)
{
    __shared__ unsigned smax[128];
    if (threadIdx.x < 128) smax[threadIdx.x] = 0u;
    __syncthreads();
    int i = blockIdx.x * blockDim.x + threadIdx.x;
    bool take = false;
    float g = 0.f;
    if (i < n) {
        unsigned T = (unsigned)state[0];
        int krem = state[1];
        unsigned k = keys[i];
        if (k > T) take = true;
        else if (k == T) {
            int idx = atomicAdd(&state[2], 1);
            if (idx < krem) take = true;
        }
        if (take) g = tanhf(score[i]);
    }
    if (take) {
        const float* xr = h + (size_t)i * 128;
        for (int f = 0; f < 128; ++f) atomicMax(&smax[f], flipF(g * xr[f]));
    }
    __syncthreads();
    if (threadIdx.x < 128 && smax[threadIdx.x])
        atomicMax(agenc + threadIdx.x, smax[threadIdx.x]);
}

// ---------------- final MLP head (single block of 128 threads) ----------------
__global__ __launch_bounds__(128) void head(
    const unsigned* __restrict__ agenc, const float* __restrict__ addf,
    const float* __restrict__ fc1w, const float* __restrict__ fc1b,
    const float* __restrict__ fc1a, const float* __restrict__ fc2w1,
    const float* __restrict__ fc2b1, const float* __restrict__ fc2a,
    const float* __restrict__ fc2w2, const float* __restrict__ fc2b2,
    const float* __restrict__ fc3w, const float* __restrict__ fc3b,
    const float* __restrict__ multw, const float* __restrict__ multb,
    float* __restrict__ out)
{
    __shared__ float pool[132];
    __shared__ float x[128];
    __shared__ float y[128];
    __shared__ float red[128];
    int t = threadIdx.x;
    pool[t] = unflipF(agenc[t]);
    if (t < 4) pool[128 + t] = addf[t];
    __syncthreads();
    float s = fc1b[t];
    for (int i = 0; i < 132; ++i) s += pool[i] * fc1w[i * 128 + t];
    float a1 = fc1a[0];
    x[t] = (s >= 0.f) ? s : a1 * s;
    __syncthreads();
    float s2 = fc2b1[t];
    for (int i = 0; i < 128; ++i) s2 += x[i] * fc2w1[i * 128 + t];
    float a2 = fc2a[0];
    y[t] = (s2 >= 0.f) ? s2 : a2 * s2;
    __syncthreads();

    red[t] = y[t] * fc2w2[t];
    __syncthreads();
    for (int o = 64; o > 0; o >>= 1) { if (t < o) red[t] += red[t + o]; __syncthreads(); }
    float r1 = red[0];
    __syncthreads();
    red[t] = x[t] * fc3w[t];
    __syncthreads();
    for (int o = 64; o > 0; o >>= 1) { if (t < o) red[t] += red[t + o]; __syncthreads(); }
    float r2 = red[0];
    __syncthreads();
    red[t] = x[t] * multw[t];
    __syncthreads();
    for (int o = 64; o > 0; o >>= 1) { if (t < o) red[t] += red[t + o]; __syncthreads(); }
    float r3 = red[0];
    if (t == 0) {
        out[0] = expf(r1 + fc2b2[0]);
        out[1] = expf(r2 + fc3b[0]) * tanhf(r3 + multb[0]);
    }
}

// =============================== launcher ===============================
extern "C" void kernel_launch(void* const* d_in, const int* in_sizes, int n_in,
                              void* d_out, int out_size, void* d_ws, size_t ws_size,
                              hipStream_t stream)
{
    (void)in_sizes; (void)n_in; (void)out_size; (void)ws_size;
    const float* Ax = (const float*)d_in[0];
    const float* Gx = (const float*)d_in[1];
    const float* addf = (const float*)d_in[2];
    const float* gat_w = (const float*)d_in[3];
    const float* att_src = (const float*)d_in[4];
    const float* att_dst = (const float*)d_in[5];
    const float* gat_bias = (const float*)d_in[6];
    const float* bn1_w = (const float*)d_in[7];
    const float* bn1_b = (const float*)d_in[8];
    const float* prelu1_a = (const float*)d_in[9];
    const float* sage_wl = (const float*)d_in[10];
    const float* sage_bl = (const float*)d_in[11];
    const float* sage_wr = (const float*)d_in[12];
    const float* bn2_w = (const float*)d_in[13];
    const float* bn2_b = (const float*)d_in[14];
    const float* prelu2_a = (const float*)d_in[15];
    const float* pool_wrel = (const float*)d_in[16];
    const float* pool_brel = (const float*)d_in[17];
    const float* pool_wroot = (const float*)d_in[18];
    const float* fc1_w = (const float*)d_in[19];
    const float* fc1_b = (const float*)d_in[20];
    const float* fc1_a = (const float*)d_in[21];
    const float* fc2_w1 = (const float*)d_in[22];
    const float* fc2_b1 = (const float*)d_in[23];
    const float* fc2_a = (const float*)d_in[24];
    const float* fc2_w2 = (const float*)d_in[25];
    const float* fc2_b2 = (const float*)d_in[26];
    const float* fc3_w = (const float*)d_in[27];
    const float* fc3_b = (const float*)d_in[28];
    const float* mult_w = (const float*)d_in[29];
    const float* mult_b = (const float*)d_in[30];
    const int* Aei = (const int*)d_in[31];
    const int* Gei = (const int*)d_in[32];

    const size_t NH = (size_t)Nn * Hh;
    float* ws   = (float*)d_ws;
    float* xw   = ws;                 // N*H
    float* h    = xw + NH;            // N*H
    float* agg  = h + NH;             // N*H (also used as encoded-uint buffer)
    __bf16* gatp = (__bf16*)(agg + NH);         // 16384 bf16 (8192 floats), 32B-aligned
    __bf16* wlp  = (__bf16*)(agg + NH + 8192);  // 16384 bf16
    __bf16* wrp  = (__bf16*)(agg + NH + 16384); // 16384 bf16
    float* ssrc = agg + NH + 24576;   // N
    float* sdst = ssrc + Nn;          // N
    float* mrow = sdst + Nn;          // N (encoded, then decoded in place)
    float* denom = mrow + Nn;         // N
    float* score = denom + Nn;        // N
    unsigned* keys = (unsigned*)(score + Nn);   // N
    float* bnsum = (float*)(keys + Nn);         // 128
    float* bnsq  = bnsum + Hh;                  // 128
    unsigned* agenc = (unsigned*)(bnsq + Hh);   // 128
    unsigned* hist  = agenc + Hh;               // 256
    int* state = (int*)(hist + 256);            // 4

    const int TB = 256;
    const int Etot = Ee + Nn;
    const unsigned gN   = (Nn + TB - 1) / TB;
    const unsigned gNH  = (unsigned)((NH + TB - 1) / TB);
    const unsigned gE   = (Etot + TB - 1) / TB;
    const unsigned gE32 = (unsigned)(((long)Etot * 32 + TB - 1) / TB);
    const unsigned gF32 = (unsigned)(((long)Ee * 32 + TB - 1) / TB);
    const unsigned gGemm = (Nn + 63) / 64;

    // pack weights into WMMA B-fragment bf16 order (shared by both towers)
    pack_b<<<4, 256, 0, stream>>>(gat_w, gatp);
    pack_b<<<4, 256, 0, stream>>>(sage_wl, wlp);
    pack_b<<<4, 256, 0, stream>>>(sage_wr, wrp);

    fill_u32<<<1, 128, 0, stream>>>(agenc, 0u, Hh);

    for (int tower = 0; tower < 2; ++tower) {
        const float* X = tower ? Gx : Ax;
        const int* src = tower ? Gei : Aei;
        const int* dst = src + Ee;

        // --- GAT ---
        gemm_h128<<<gGemm, 256, 0, stream>>>(X, gatp, nullptr, nullptr, nullptr, xw, Nn);
        gat_proj<<<(Nn + 7) / 8, 256, 0, stream>>>(xw, att_src, att_dst, ssrc, sdst, Nn);
        fill_u32<<<gN, TB, 0, stream>>>((unsigned*)mrow, 0u, Nn);
        fill_f32<<<gN, TB, 0, stream>>>(denom, 0.f, Nn);
        gat_edge_max<<<gE, TB, 0, stream>>>(src, dst, Ee, Nn, ssrc, sdst, (unsigned*)mrow);
        unflip_inplace<<<gN, TB, 0, stream>>>((unsigned*)mrow, Nn);
        gat_edge_den<<<gE, TB, 0, stream>>>(src, dst, Ee, Nn, ssrc, sdst, mrow, denom);
        fill_u32<<<gNH, TB, 0, stream>>>((unsigned*)agg, 0u, NH);
        gat_edge_aggmax<<<gE32, TB, 0, stream>>>(src, dst, Ee, Nn, ssrc, sdst, mrow, denom,
                                                 xw, (unsigned*)agg);
        decode_bias<<<gNH, TB, 0, stream>>>((unsigned*)agg, gat_bias, h, NH);

        // --- BN1 + PReLU ---
        fill_f32<<<1, 256, 0, stream>>>(bnsum, 0.f, 2 * Hh);
        bn_stats<<<(Nn + 255) / 256, 128, 0, stream>>>(h, Nn, bnsum, bnsq);
        bn_prelu<<<gNH, TB, 0, stream>>>(h, h, Nn, bnsum, bnsq, bn1_w, bn1_b, prelu1_a);

        // --- SAGE (max aggr) ---
        fill_u32<<<gNH, TB, 0, stream>>>((unsigned*)agg, 0u, NH);
        edge_feat_max<<<gF32, TB, 0, stream>>>(src, dst, Ee, h, (unsigned*)agg);
        sage_decode<<<gNH, TB, 0, stream>>>((unsigned*)agg, NH);
        gemm_h128<<<gGemm, 256, 0, stream>>>(agg, wlp, h, wrp, sage_bl, xw, Nn);

        // --- BN2 + PReLU ---
        fill_f32<<<1, 256, 0, stream>>>(bnsum, 0.f, 2 * Hh);
        bn_stats<<<(Nn + 255) / 256, 128, 0, stream>>>(xw, Nn, bnsum, bnsq);
        bn_prelu<<<gNH, TB, 0, stream>>>(xw, xw, Nn, bnsum, bnsq, bn2_w, bn2_b, prelu2_a);

        // --- SAGPool: score = GraphConv(add) ---
        fill_f32<<<gNH, TB, 0, stream>>>(agg, 0.f, NH);
        edge_feat_sum<<<gF32, TB, 0, stream>>>(src, dst, Ee, xw, agg);
        pool_score<<<(Nn + 7) / 8, 256, 0, stream>>>(agg, xw, pool_wrel, pool_wroot,
                                                     pool_brel, score, keys, Nn);

        // --- top-K threshold via 4-pass radix select ---
        init_state<<<1, 1, 0, stream>>>(state, Kk);
        for (int p = 0; p < 4; ++p) {
            int shift = 24 - 8 * p;
            fill_u32<<<1, 256, 0, stream>>>(hist, 0u, 256);
            radix_hist<<<gN, TB, 0, stream>>>(keys, Nn, state, shift, hist);
            radix_select<<<1, 1, 0, stream>>>(hist, state, shift, (p == 3) ? 1 : 0);
        }
        pool_gather<<<gN, TB, 0, stream>>>(keys, score, xw, Nn, state, agenc);
    }

    head<<<1, 128, 0, stream>>>(agenc, addf, fc1_w, fc1_b, fc1_a,
                                fc2_w1, fc2_b1, fc2_a, fc2_w2, fc2_b2,
                                fc3_w, fc3_b, mult_w, mult_b, (float*)d_out);
}
